// CapsNet_79680233275573
// MI455X (gfx1250) — matrix operand order repaired and checked by reading, not compiled, not run on hardware
//
#include <hip/hip_runtime.h>
#include <hip/hip_bf16.h>
#include <math.h>

// ---------- types ----------
typedef __attribute__((ext_vector_type(16))) __bf16 v16bf;
typedef __attribute__((ext_vector_type(8)))  float  v8f;
typedef int gv4i __attribute__((vector_size(16)));   // pointee type the async builtins expect
typedef int gv2i __attribute__((vector_size(8)));

// ---------- CDNA5 async global->LDS copy (ASYNCcnt-tracked) ----------
#if defined(__HIP_DEVICE_COMPILE__) && \
    __has_builtin(__builtin_amdgcn_global_load_async_to_lds_b128) && \
    __has_builtin(__builtin_amdgcn_global_load_async_to_lds_b64) && \
    __has_builtin(__builtin_amdgcn_s_wait_asynccnt)
#define HAVE_ASYNC_LDS 1
__device__ __forceinline__ void async_cp_b128(const void* g, void* l) {
  __builtin_amdgcn_global_load_async_to_lds_b128(
      (__attribute__((address_space(1))) gv4i*)(g),
      (__attribute__((address_space(3))) gv4i*)(l), 0, 0);
}
__device__ __forceinline__ void async_cp_b64(const void* g, void* l) {
  __builtin_amdgcn_global_load_async_to_lds_b64(
      (__attribute__((address_space(1))) gv2i*)(g),
      (__attribute__((address_space(3))) gv2i*)(l), 0, 0);
}
#define WAIT_ASYNC(n) __builtin_amdgcn_s_wait_asynccnt(n)
#else
#define WAIT_ASYNC(n) ((void)0)
#endif

__device__ __forceinline__ __bf16 f2bf(float f) {
  unsigned u = __builtin_bit_cast(unsigned, f);
  u += 0x7FFFu + ((u >> 16) & 1u);            // round-to-nearest-even
  unsigned short h = (unsigned short)(u >> 16);
  return __builtin_bit_cast(__bf16, h);
}

// ---------- WMMA fragment loaders (CDNA5 16-bit layouts, wave32) ----------
__device__ __forceinline__ v16bf load_a_frag(const __bf16* As, int lane, int ld) {
  int m = lane & 15;
  int h = lane >> 4;
  v16bf a;
#pragma unroll
  for (int v = 0; v < 8; ++v) {
    int k0 = (h << 4) + (v << 1);
    a[2 * v]     = As[m * ld + k0];
    a[2 * v + 1] = As[m * ld + k0 + 1];
  }
  return a;
}
// B fragment from pre-swizzled storage: [lane][16 contiguous bf16]
__device__ __forceinline__ v16bf load_b_swz(const __bf16* Bs, int lane) {
  return *(const v16bf*)(Bs + lane * 16);
}

// swizzled B index: element (k,n) of a [K][N] matrix ->
//   (((k/32)*NT + n/16)*32 + (k%32)/16*16 + n%16)*16 + k%16

// ---------- pack kernels (dst-indexed; contiguous writes) ----------
__global__ void pack_x_k(const float* __restrict__ x, __bf16* __restrict__ xb, int total) {
  int idx = blockIdx.x * blockDim.x + threadIdx.x;
  if (idx >= total) return;
  int c = idx & 63;
  int rest = idx >> 6;          // b*484 + s
  int s = rest % 484;
  int b = rest / 484;
  xb[idx] = f2bf(x[((size_t)b * 64 + c) * 484 + s]);
}
__global__ void pack_w1_swz(const float* __restrict__ w, __bf16* __restrict__ o, int total) {
  int idx = blockIdx.x * blockDim.x + threadIdx.x;
  if (idx >= total) return;
  int e  = idx & 15;
  int l  = (idx >> 4) & 31;
  int rest = idx >> 9;          // kc*16 + nt
  int nt = rest & 15;
  int kc = rest >> 4;
  int k  = kc * 32 + (l >> 4) * 16 + e;
  int n  = nt * 16 + (l & 15);
  int c  = k & 63, r = k >> 6;  // r = kh*9+kw
  o[idx] = f2bf(w[(size_t)n * 5184 + c * 81 + r]);
}
__global__ void pack_w2_swz(const float* __restrict__ w, __bf16* __restrict__ o, int total) {
  int idx = blockIdx.x * blockDim.x + threadIdx.x;
  if (idx >= total) return;
  int e  = idx & 15;
  int l  = (idx >> 4) & 31;
  int rest = idx >> 9;          // kc*3 + nt
  int nt = rest % 3;
  int kc = rest / 3;
  int k  = kc * 32 + (l >> 4) * 16 + e;
  int n  = nt * 16 + (l & 15);
  int c  = k & 255, r = k >> 8; // r = kh*8+kw
  o[idx] = f2bf(w[(size_t)n * 16384 + c * 64 + r]);
}
__global__ void pack_dec_swz(const float* __restrict__ w, __bf16* __restrict__ o, int N, int K) {
  int idx = blockIdx.x * blockDim.x + threadIdx.x;
  if (idx >= N * K) return;
  int e  = idx & 15;
  int l  = (idx >> 4) & 31;
  int rest = idx >> 9;          // kc*NT + nt
  int NT = N >> 4;
  int nt = rest % NT;
  int kc = rest / NT;
  int k  = kc * 32 + (l >> 4) * 16 + e;
  int n  = nt * 16 + (l & 15);
  o[idx] = f2bf(w[(size_t)n * K + k]);
}

// ---------- conv1: implicit GEMM, M=401408 N=256 K=5184 ----------
__global__ __launch_bounds__(128) void conv1_wmma(
    const __bf16* __restrict__ xb,   // [B][484][64]
    const __bf16* __restrict__ w1s,  // swizzled weights, NT=16
    const float*  __restrict__ bias, // [256]
    __bf16* __restrict__ out1)       // [401408][256]  (== NHWC activation)
{
  __shared__ __attribute__((aligned(16))) __bf16 As[2][64 * 32];
  __shared__ __attribute__((aligned(16))) __bf16 Bs[2][2048];
  const int tid  = threadIdx.x;
  const int lane = tid & 31, wave = tid >> 5;
  const int m0  = blockIdx.x * 64;
  const int nt0 = blockIdx.y * 4;

  const int arow  = tid >> 1;
  const int acoff = (tid & 1) * 16;
  int m = m0 + arow;
  int b = m / 196;
  int p = m - b * 196;
  int oh = p / 14;
  int ow = p - oh * 14;
  const __bf16* arow_base = xb + ((size_t)b * 484 + oh * 22 + ow) * 64;

  auto stage = [&](int kc, int buf) {
    int r  = kc >> 1;
    int c0 = (kc & 1) * 32;
    int kh = r / 9, kw = r - kh * 9;
    const __bf16* sA = arow_base + (kh * 22 + kw) * 64 + c0 + acoff;
    __bf16* dA = &As[buf][arow * 32 + acoff];
    const __bf16* sB = w1s + ((size_t)kc * 16 + nt0) * 512 + tid * 16;
    __bf16* dB = &Bs[buf][tid * 16];
#ifdef HAVE_ASYNC_LDS
    async_cp_b128(sA, dA);  async_cp_b128(sA + 8, dA + 8);
    async_cp_b128(sB, dB);  async_cp_b128(sB + 8, dB + 8);
#else
    ((uint4*)dA)[0] = ((const uint4*)sA)[0];
    ((uint4*)dA)[1] = ((const uint4*)sA)[1];
    ((uint4*)dB)[0] = ((const uint4*)sB)[0];
    ((uint4*)dB)[1] = ((const uint4*)sB)[1];
#endif
  };

  v8f acc[4];
#pragma unroll
  for (int t = 0; t < 4; ++t) acc[t] = (v8f){0, 0, 0, 0, 0, 0, 0, 0};

  auto mma = [&](const __bf16* As_, const __bf16* Bs_) {
    v16bf a = load_a_frag(As_ + wave * 16 * 32, lane, 32);
#pragma unroll
    for (int t = 0; t < 4; ++t) {
      v16bf bf = load_b_swz(Bs_ + t * 512, lane);
      acc[t] = __builtin_amdgcn_wmma_f32_16x16x32_bf16(false, a, false, bf,
                                                       (short)0, acc[t], false, false);
    }
  };

  stage(0, 0);
  for (int kc = 0; kc + 1 < 162; ++kc) {     // steady state: branch-free body
    stage(kc + 1, (kc + 1) & 1);
    WAIT_ASYNC(4);
    __syncthreads();
    mma(As[kc & 1], Bs[kc & 1]);
    __syncthreads();
  }
  WAIT_ASYNC(0);                              // peeled tail
  __syncthreads();
  mma(As[161 & 1], Bs[161 & 1]);

  const int mrow = m0 + wave * 16 + (lane >> 4) * 8;
  const int nlo  = lane & 15;
#pragma unroll
  for (int t = 0; t < 4; ++t) {
    int n = nt0 * 16 + t * 16 + nlo;
    float bv = bias[n];
#pragma unroll
    for (int r2 = 0; r2 < 8; ++r2) {
      float v = fmaxf(acc[t][r2] + bv, 0.f);   // ReLU
      out1[(size_t)(mrow + r2) * 256 + n] = f2bf(v);
    }
  }
}

// ---------- conv2 (primary caps): M=32768 N=48 K=16384, stride 2 ----------
__global__ __launch_bounds__(128) void conv2_wmma(
    const __bf16* __restrict__ out1, // [b][14][14][256]
    const __bf16* __restrict__ w2s,  // swizzled weights, NT=3
    const float*  __restrict__ bias, // [48]
    float* __restrict__ pc)          // [2048][48][4][4] NCHW f32
{
  __shared__ __attribute__((aligned(16))) __bf16 As[2][64 * 32];
  __shared__ __attribute__((aligned(16))) __bf16 Bs[2][1536];
  const int tid  = threadIdx.x;
  const int lane = tid & 31, wave = tid >> 5;
  const int m0 = blockIdx.x * 64;

  const int arow  = tid >> 1;
  const int acoff = (tid & 1) * 16;
  int m  = m0 + arow;
  int b  = m >> 4;
  int q  = m & 15;
  int oh = q >> 2, ow = q & 3;
  const __bf16* arow_base = out1 + ((size_t)b * 196 + (2 * oh) * 14 + 2 * ow) * 256;

  auto stage = [&](int kc, int buf) {
    int r  = kc >> 3;
    int c0 = (kc & 7) * 32;
    int kh = r >> 3, kw = r & 7;
    const __bf16* sA = arow_base + (kh * 14 + kw) * 256 + c0 + acoff;
    __bf16* dA = &As[buf][arow * 32 + acoff];
    const __bf16* sB = w2s + (size_t)kc * 1536 + tid * 12;   // 12 bf16 = 24B/thread
    __bf16* dB = &Bs[buf][tid * 12];
#ifdef HAVE_ASYNC_LDS
    async_cp_b128(sA, dA);      async_cp_b128(sA + 8, dA + 8);
    async_cp_b64(sB, dB);       async_cp_b64(sB + 4, dB + 4);
    async_cp_b64(sB + 8, dB + 8);
#else
    ((uint4*)dA)[0] = ((const uint4*)sA)[0];
    ((uint4*)dA)[1] = ((const uint4*)sA)[1];
    ((uint2*)dB)[0] = ((const uint2*)sB)[0];
    ((uint2*)dB)[1] = ((const uint2*)sB)[1];
    ((uint2*)dB)[2] = ((const uint2*)sB)[2];
#endif
  };

  v8f acc[3];
#pragma unroll
  for (int t = 0; t < 3; ++t) acc[t] = (v8f){0, 0, 0, 0, 0, 0, 0, 0};

  auto mma = [&](const __bf16* As_, const __bf16* Bs_) {
    v16bf a = load_a_frag(As_ + wave * 16 * 32, lane, 32);
#pragma unroll
    for (int t = 0; t < 3; ++t) {
      v16bf bf = load_b_swz(Bs_ + t * 512, lane);
      acc[t] = __builtin_amdgcn_wmma_f32_16x16x32_bf16(false, a, false, bf,
                                                       (short)0, acc[t], false, false);
    }
  };

  stage(0, 0);
  for (int kc = 0; kc + 1 < 512; ++kc) {
    stage(kc + 1, (kc + 1) & 1);
    WAIT_ASYNC(5);
    __syncthreads();
    mma(As[kc & 1], Bs[kc & 1]);
    __syncthreads();
  }
  WAIT_ASYNC(0);
  __syncthreads();
  mma(As[511 & 1], Bs[511 & 1]);

  const int mrow = m0 + wave * 16 + (lane >> 4) * 8;
  const int nlo  = lane & 15;
#pragma unroll
  for (int t = 0; t < 3; ++t) {
    int ch = t * 16 + nlo;            // 0..47
    float bv = bias[ch];
#pragma unroll
    for (int r2 = 0; r2 < 8; ++r2) {
      int mm = mrow + r2;
      int b2 = mm >> 4, q2 = mm & 15;
      pc[((size_t)b2 * 48 + ch) * 16 + q2] = acc[t][r2] + bv;  // no ReLU
    }
  }
}

// ---------- capsule squash + 2-iteration routing ----------
__global__ __launch_bounds__(128) void caps_route(
    const float* __restrict__ pc,   // [B][768]
    const float* __restrict__ Wd,   // [2][96][16][8]
    float* __restrict__ logits,     // [B][2]
    __bf16* __restrict__ h)         // [B][32]
{
  __shared__ float pcL[768];
  __shared__ float uL[768];
  __shared__ float uhatL[3072];     // [o][i][d]
  __shared__ float bL[192];
  __shared__ float cL[192];
  __shared__ float sL[32];
  __shared__ float vL[32];
  __shared__ float scaleL[2];
  __shared__ float logitL[2];
  __shared__ int   amL;
  const int tid = threadIdx.x;
  const int b   = blockIdx.x;

  const float* pcg = pc + (size_t)b * 768;
  for (int j = tid; j < 768; j += 128) pcL[j] = pcg[j];
  __syncthreads();

  if (tid < 96) {                           // squash each 8-vector capsule
    float sq = 0.f;
#pragma unroll
    for (int k = 0; k < 8; ++k) { float x = pcL[tid * 8 + k]; sq += x * x; }
    float n  = sqrtf(sq);
    float sc = tanhf(n) / (n + 1e-8f);
#pragma unroll
    for (int k = 0; k < 8; ++k) uL[tid * 8 + k] = sc * pcL[tid * 8 + k];
  }
  __syncthreads();

  for (int idx = tid; idx < 3072; idx += 128) {   // u_hat
    int i = (idx % 1536) >> 4;
    const float* w = Wd + (size_t)idx * 8;
    float s = 0.f;
#pragma unroll
    for (int k = 0; k < 8; ++k) s += w[k] * uL[i * 8 + k];
    uhatL[idx] = s;
  }
  __syncthreads();

  if (tid < 32) {                           // iter 1: c = softmax(0) = 0.5
    int o = tid >> 4, d = tid & 15;
    float s = 0.f;
    for (int i = 0; i < 96; ++i) s += uhatL[o * 1536 + i * 16 + d];
    sL[tid] = 0.5f * s;
  }
  __syncthreads();
  if (tid < 2) {
    float sq = 0.f;
    for (int d = 0; d < 16; ++d) { float x = sL[tid * 16 + d]; sq += x * x; }
    float n = sqrtf(sq);
    scaleL[tid] = tanhf(n) / (n + 1e-8f);
  }
  __syncthreads();
  if (tid < 32) vL[tid] = scaleL[tid >> 4] * sL[tid];
  __syncthreads();

  for (int idx = tid; idx < 192; idx += 128) {    // b[o][i] = <u_hat, v>
    int o = idx / 96, i = idx - o * 96;
    float s = 0.f;
#pragma unroll
    for (int d = 0; d < 16; ++d) s += uhatL[o * 1536 + i * 16 + d] * vL[o * 16 + d];
    bL[idx] = s;
  }
  __syncthreads();

  if (tid < 96) {                           // softmax over o (2 classes)
    float e0 = expf(bL[tid]), e1 = expf(bL[96 + tid]);
    float inv = 1.f / (e0 + e1);
    cL[tid] = e0 * inv;
    cL[96 + tid] = e1 * inv;
  }
  __syncthreads();

  if (tid < 32) {
    int o = tid >> 4, d = tid & 15;
    float s = 0.f;
    for (int i = 0; i < 96; ++i) s += cL[o * 96 + i] * uhatL[o * 1536 + i * 16 + d];
    sL[tid] = s;
  }
  __syncthreads();
  if (tid < 2) {
    float sq = 0.f;
    for (int d = 0; d < 16; ++d) { float x = sL[tid * 16 + d]; sq += x * x; }
    float n  = sqrtf(sq);
    float sc = tanhf(n) / (n + 1e-8f);
    scaleL[tid] = sc;
    float lg = sc * n;                      // ||squash(s)||
    logitL[tid] = lg;
    logits[(size_t)b * 2 + tid] = lg;
  }
  __syncthreads();
  if (tid == 0) amL = (logitL[1] > logitL[0]) ? 1 : 0;
  __syncthreads();
  if (tid < 32) {
    int o = tid >> 4;
    float v = (o == amL) ? scaleL[o] * sL[tid] : 0.f;
    h[(size_t)b * 32 + tid] = f2bf(v);
  }
}

// ---------- generic decoder GEMM: out = act(A[M][K] * Wswz + bias) ----------
__global__ __launch_bounds__(128) void gemm_act(
    const __bf16* __restrict__ A,   // [M][K]
    const __bf16* __restrict__ Ws,  // swizzled [K][N]
    const float*  __restrict__ bias,
    __bf16* __restrict__ obf,       // act==0: ReLU -> bf16
    float*  __restrict__ of32,      // act==1: sigmoid -> f32
    int K, int N, int act)
{
  __shared__ __attribute__((aligned(16))) __bf16 As[2][64 * 32];
  __shared__ __attribute__((aligned(16))) __bf16 Bs[2][2048];
  const int tid  = threadIdx.x;
  const int lane = tid & 31, wave = tid >> 5;
  const int m0  = blockIdx.x * 64;
  const int nt0 = blockIdx.y * 4;
  const int NT  = N >> 4;

  const int arow  = tid >> 1;
  const int acoff = (tid & 1) * 16;

  auto stage = [&](int kc, int buf) {
    const __bf16* sA = A + (size_t)(m0 + arow) * K + kc * 32 + acoff;
    __bf16* dA = &As[buf][arow * 32 + acoff];
    const __bf16* sB = Ws + ((size_t)kc * NT + nt0) * 512 + tid * 16;
    __bf16* dB = &Bs[buf][tid * 16];
#ifdef HAVE_ASYNC_LDS
    async_cp_b128(sA, dA);  async_cp_b128(sA + 8, dA + 8);
    async_cp_b128(sB, dB);  async_cp_b128(sB + 8, dB + 8);
#else
    ((uint4*)dA)[0] = ((const uint4*)sA)[0];
    ((uint4*)dA)[1] = ((const uint4*)sA)[1];
    ((uint4*)dB)[0] = ((const uint4*)sB)[0];
    ((uint4*)dB)[1] = ((const uint4*)sB)[1];
#endif
  };

  v8f acc[4];
#pragma unroll
  for (int t = 0; t < 4; ++t) acc[t] = (v8f){0, 0, 0, 0, 0, 0, 0, 0};

  auto mma = [&](const __bf16* As_, const __bf16* Bs_) {
    v16bf a = load_a_frag(As_ + wave * 16 * 32, lane, 32);
#pragma unroll
    for (int t = 0; t < 4; ++t) {
      v16bf bf = load_b_swz(Bs_ + t * 512, lane);
      acc[t] = __builtin_amdgcn_wmma_f32_16x16x32_bf16(false, a, false, bf,
                                                       (short)0, acc[t], false, false);
    }
  };

  const int nk = K >> 5;
  stage(0, 0);
  for (int kc = 0; kc + 1 < nk; ++kc) {
    stage(kc + 1, (kc + 1) & 1);
    WAIT_ASYNC(4);
    __syncthreads();
    mma(As[kc & 1], Bs[kc & 1]);
    __syncthreads();
  }
  WAIT_ASYNC(0);
  __syncthreads();
  mma(As[(nk - 1) & 1], Bs[(nk - 1) & 1]);

  const int mrow = m0 + wave * 16 + (lane >> 4) * 8;
  const int nlo  = lane & 15;
#pragma unroll
  for (int t = 0; t < 4; ++t) {
    int n = nt0 * 16 + t * 16 + nlo;
    float bv = bias[n];
#pragma unroll
    for (int r2 = 0; r2 < 8; ++r2) {
      float v = acc[t][r2] + bv;
      if (act == 0) {
        obf[(size_t)(mrow + r2) * N + n] = f2bf(fmaxf(v, 0.f));
      } else {
        of32[(size_t)(mrow + r2) * N + n] = 1.f / (1.f + expf(-v));
      }
    }
  }
}

// ---------- host launch ----------
extern "C" void kernel_launch(void* const* d_in, const int* in_sizes, int n_in,
                              void* d_out, int out_size, void* d_ws, size_t ws_size,
                              hipStream_t stream) {
  (void)in_sizes; (void)n_in; (void)out_size; (void)ws_size;
  const float* x   = (const float*)d_in[0];
  const float* w1  = (const float*)d_in[1];
  const float* b1  = (const float*)d_in[2];
  const float* w2  = (const float*)d_in[3];
  const float* b2  = (const float*)d_in[4];
  const float* Wd  = (const float*)d_in[5];
  const float* dw1 = (const float*)d_in[6];
  const float* db1 = (const float*)d_in[7];
  const float* dw2 = (const float*)d_in[8];
  const float* db2 = (const float*)d_in[9];
  const float* dw3 = (const float*)d_in[10];
  const float* db3 = (const float*)d_in[11];
  float* out = (float*)d_out;

  char* ws = (char*)d_ws;
  size_t off = 0;
  auto alloc = [&](size_t bytes) -> void* {
    void* p = ws + off;
    off = (off + bytes + 255) & ~(size_t)255;
    return p;
  };
  const size_t XB_E = (size_t)2048 * 484 * 64;
  const size_t W1_E = (size_t)5184 * 256;
  const size_t O1_E = (size_t)401408 * 256;
  const size_t W2_E = (size_t)16384 * 48;
  const size_t PC_E = (size_t)2048 * 768;
  __bf16* xb   = (__bf16*)alloc(XB_E * 2);
  __bf16* w1s  = (__bf16*)alloc(W1_E * 2);
  __bf16* out1 = (__bf16*)alloc(O1_E * 2);
  __bf16* w2s  = (__bf16*)alloc(W2_E * 2);
  float*  pcb  = (float*)alloc(PC_E * 4);
  __bf16* hB   = (__bf16*)alloc((size_t)2048 * 32 * 2);
  __bf16* w1d  = (__bf16*)alloc((size_t)32 * 512 * 2);
  __bf16* h1B  = (__bf16*)alloc((size_t)2048 * 512 * 2);
  __bf16* w2d  = (__bf16*)alloc((size_t)512 * 1024 * 2);
  __bf16* h2B  = (__bf16*)alloc((size_t)2048 * 1024 * 2);
  __bf16* w3d  = (__bf16*)alloc((size_t)1024 * 256 * 2);

  pack_x_k   <<<(unsigned)((XB_E + 255) / 256), 256, 0, stream>>>(x, xb, (int)XB_E);
  pack_w1_swz<<<(unsigned)((W1_E + 255) / 256), 256, 0, stream>>>(w1, w1s, (int)W1_E);
  pack_w2_swz<<<(unsigned)((W2_E + 255) / 256), 256, 0, stream>>>(w2, w2s, (int)W2_E);
  pack_dec_swz<<<(32 * 512 + 255) / 256, 256, 0, stream>>>(dw1, w1d, 512, 32);
  pack_dec_swz<<<(512 * 1024 + 255) / 256, 256, 0, stream>>>(dw2, w2d, 1024, 512);
  pack_dec_swz<<<(1024 * 256 + 255) / 256, 256, 0, stream>>>(dw3, w3d, 256, 1024);

  conv1_wmma<<<dim3(6272, 4), 128, 0, stream>>>(xb, w1s, b1, out1);
  conv2_wmma<<<512, 128, 0, stream>>>(out1, w2s, b2, pcb);
  caps_route<<<2048, 128, 0, stream>>>(pcb, Wd, out, hB);
  gemm_act<<<dim3(32, 8),  128, 0, stream>>>(hB,  w1d, db1, h1B, nullptr, 32,   512,  0);
  gemm_act<<<dim3(32, 16), 128, 0, stream>>>(h1B, w2d, db2, h2B, nullptr, 512,  1024, 0);
  gemm_act<<<dim3(32, 4),  128, 0, stream>>>(h2B, w3d, db3, nullptr, out + 4096, 1024, 256, 1);
}
